// DynamicExpertGating_20109036880143
// MI455X (gfx1250) — matrix operand order repaired and checked by reading, not compiled, not run on hardware
//
#include <hip/hip_runtime.h>
#include <math.h>

typedef __attribute__((ext_vector_type(16))) __bf16 v16bf;
typedef __attribute__((ext_vector_type(8)))  float  v8f;
typedef __attribute__((__vector_size__(4 * sizeof(int)))) int v4i;

#define NTOK   8192      // B*S
#define H_DIM  1024
#define D_DIM  1024
#define NEXP   8
#define BM     64        // block M tile
#define TILE_N 128       // block N tile
#define KSTEP  32
#define LDA    40        // padded LDS row stride (bf16 elems) = 80B, 16B-aligned rows
#define LDB    40

#if __has_builtin(__builtin_amdgcn_global_load_async_to_lds_b128)
#define HAS_ASYNC_LDS 1
#else
#define HAS_ASYNC_LDS 0
#endif

// 16B global -> LDS copy: async DMA (ASYNCcnt) when available, else via VGPRs
__device__ __forceinline__ void cp16_g2l(const void* gsrc, void* ldst) {
#if HAS_ASYNC_LDS
  __builtin_amdgcn_global_load_async_to_lds_b128(
      (__attribute__((address_space(1))) v4i*)gsrc,
      (__attribute__((address_space(3))) v4i*)ldst, 0, 0);
#else
  *(uint4*)ldst = *(const uint4*)gsrc;
#endif
}

__device__ __forceinline__ void cp16_wait() {
#if HAS_ASYNC_LDS
#if __has_builtin(__builtin_amdgcn_s_wait_asynccnt)
  __builtin_amdgcn_s_wait_asynccnt(0);
#else
  asm volatile("s_wait_asynccnt 0x0" ::: "memory");
#endif
#endif
}

__device__ __forceinline__ float gelu_tanh(float x) {
  // 0.5*x*(1+tanh(u)) == x * sigmoid(2u)
  float u = 0.7978845608028654f * (x + 0.044715f * x * x * x);
  u = fminf(fmaxf(u, -15.f), 15.f);
  return x / (1.f + __expf(-2.f * u));
}

__device__ __forceinline__ unsigned pack_bf16(float a, float b) {
  unsigned lo = (unsigned)__builtin_bit_cast(unsigned short, (__bf16)a);
  unsigned hi = (unsigned)__builtin_bit_cast(unsigned short, (__bf16)b);
  return lo | (hi << 16);
}

// ---------------- elementwise f32 -> bf16 (packed dword) ----------------------
__global__ __launch_bounds__(256) void cvt_x_kernel(const float* __restrict__ src,
                                                    unsigned* __restrict__ dst, int n2) {
  int i = blockIdx.x * 256 + threadIdx.x;
  if (i < n2) {
    float2 v = ((const float2*)src)[i];
    dst[i] = pack_bf16(v.x, v.y);
  }
}

// ---------------- tiled transpose + cvt: src[R][C] f32 -> dst[C][R] bf16 ------
__global__ __launch_bounds__(256) void transpose_cvt_kernel(
    const float* __restrict__ src, __bf16* __restrict__ dst, int R, int C) {
  __shared__ __bf16 tile[32][33];
  size_t boff = (size_t)blockIdx.z * R * C;
  src += boff;
  dst += boff;
  int c0 = blockIdx.x * 32, r0 = blockIdx.y * 32;
  int tx = threadIdx.x & 31, ty = threadIdx.x >> 5;   // 32 x 8
  #pragma unroll
  for (int j = 0; j < 4; j++) {
    int r = ty + j * 8;
    tile[r][tx] = (__bf16)src[(size_t)(r0 + r) * C + c0 + tx];
  }
  __syncthreads();
  #pragma unroll
  for (int j = 0; j < 4; j++) {
    int r = ty + j * 8;
    dst[(size_t)(c0 + r) * R + r0 + tx] = tile[tx][r];
  }
}

// ---------------- Router: logits -> softmax -> top2 -> gates + expert lists ---
__global__ __launch_bounds__(256) void router_kernel(
    const float* __restrict__ x, const float* __restrict__ w_router,
    const float* __restrict__ b_router,
    float* __restrict__ gates, int* __restrict__ counts, int* __restrict__ lists) {
  int wave  = threadIdx.x >> 5;
  int lane  = threadIdx.x & 31;
  int token = blockIdx.x * 8 + wave;
  if (token >= NTOK) return;
  const float* xr = x + (size_t)token * H_DIM;

  float acc[NEXP];
  #pragma unroll
  for (int e = 0; e < NEXP; e++) acc[e] = 0.f;
  for (int h = lane; h < H_DIM; h += 32) {
    float xv = xr[h];
    const float* wr = w_router + h * NEXP;
    #pragma unroll
    for (int e = 0; e < NEXP; e++) acc[e] += xv * wr[e];
  }
  #pragma unroll
  for (int e = 0; e < NEXP; e++)
    #pragma unroll
    for (int off = 16; off; off >>= 1)
      acc[e] += __shfl_xor(acc[e], off, 32);

  if (lane == 0) {
    float l[NEXP], m = -1e30f;
    #pragma unroll
    for (int e = 0; e < NEXP; e++) { l[e] = acc[e] + b_router[e]; m = fmaxf(m, l[e]); }
    float p[NEXP];
    #pragma unroll
    for (int e = 0; e < NEXP; e++) p[e] = __expf(l[e] - m);
    int i0 = 0;
    #pragma unroll
    for (int e = 1; e < NEXP; e++) if (p[e] > p[i0]) i0 = e;
    int i1 = (i0 == 0) ? 1 : 0;
    #pragma unroll
    for (int e = 0; e < NEXP; e++) if (e != i0 && p[e] > p[i1]) i1 = e;
    float denom = p[i0] + p[i1];
    gates[token * 2 + 0] = p[i0] / denom;
    gates[token * 2 + 1] = p[i1] / denom;
    int pos0 = atomicAdd(&counts[i0], 1);
    lists[i0 * NTOK + pos0] = token * 2 + 0;
    int pos1 = atomicAdd(&counts[i1], 1);
    lists[i1 * NTOK + pos1] = token * 2 + 1;
  }
}

// ---------------- Per-expert gathered GEMM + gelu + gated scatter-add ---------
__global__ __launch_bounds__(256) void expert_kernel(
    const __bf16* __restrict__ xbf, const __bf16* __restrict__ wT,
    const float* __restrict__ b_experts, const float* __restrict__ gates,
    const int* __restrict__ counts, const int* __restrict__ lists,
    float* __restrict__ combined) {
  __shared__ __bf16 lds_a[2][BM * LDA];
  __shared__ __bf16 lds_b[2][TILE_N * LDB];
  __shared__ int lds_list[BM];

  int e = blockIdx.z, bn = blockIdx.x, bm = blockIdx.y;
  int cnt = counts[e];
  if (bm * BM >= cnt) return;
  int tid = threadIdx.x;

  // epilogue list (clamped -> always a valid entry; extra rows never written)
  if (tid < BM) {
    int grow = bm * BM + tid;
    lds_list[tid] = lists[e * NTOK + (grow < cnt ? grow : cnt - 1)];
  }

  const __bf16* We = wT + (size_t)e * D_DIM * H_DIM + (size_t)(bn * TILE_N) * H_DIM;

  // ---- hoisted staging descriptors (b128 granularity) ----
  int t4r = tid >> 2;            // 0..63
  int t4c = (tid & 3) * 8;       // 0,8,16,24 (bf16 elems)
  int grow_a = bm * BM + t4r;
  int entry_a = lists[e * NTOK + (grow_a < cnt ? grow_a : cnt - 1)];
  const __bf16* aptr  = xbf + (size_t)(entry_a >> 1) * H_DIM + t4c;
  const __bf16* bptr0 = We + (size_t)t4r * H_DIM + t4c;
  const __bf16* bptr1 = We + (size_t)(t4r + 64) * H_DIM + t4c;
  unsigned aoff  = (unsigned)(t4r * LDA + t4c);
  unsigned boff0 = (unsigned)(t4r * LDB + t4c);
  unsigned boff1 = (unsigned)((t4r + 64) * LDB + t4c);

  auto stage = [&](int buf, int k0) {
    cp16_g2l(aptr + k0,  &lds_a[buf][aoff]);
    cp16_g2l(bptr0 + k0, &lds_b[buf][boff0]);
    cp16_g2l(bptr1 + k0, &lds_b[buf][boff1]);
  };

  int wave = tid >> 5, lane = tid & 31;
  int mw = wave >> 2, nw = wave & 3;
  int g = lane >> 4, nl = lane & 15;

  v8f acc[2][2] = {};
  stage(0, 0);
  cp16_wait();
  __syncthreads();

  const int NSTEPS = H_DIM / KSTEP;
  for (int s = 0; s < NSTEPS; s++) {
    int cur = s & 1;
    if (s + 1 < NSTEPS) stage(cur ^ 1, (s + 1) * KSTEP);

    const __bf16* ab = lds_a[cur];
    const __bf16* bb = lds_b[cur];
    union { v16bf v; unsigned u[8]; } A[2], Bf[2];
    #pragma unroll
    for (int mt = 0; mt < 2; mt++) {
      int m = mw * 32 + mt * 16 + nl;
      const unsigned* alo = (const unsigned*)&ab[m * LDA + g * 8];
      const unsigned* ahi = (const unsigned*)&ab[m * LDA + 16 + g * 8];
      #pragma unroll
      for (int j = 0; j < 4; j++) { A[mt].u[j] = alo[j]; A[mt].u[4 + j] = ahi[j]; }
    }
    #pragma unroll
    for (int nt = 0; nt < 2; nt++) {
      int n = nw * 32 + nt * 16 + nl;
      const unsigned* blo = (const unsigned*)&bb[n * LDB + g * 8];
      const unsigned* bhi = (const unsigned*)&bb[n * LDB + 16 + g * 8];
      #pragma unroll
      for (int j = 0; j < 4; j++) { Bf[nt].u[j] = blo[j]; Bf[nt].u[4 + j] = bhi[j]; }
    }
    #pragma unroll
    for (int mt = 0; mt < 2; mt++)
      #pragma unroll
      for (int nt = 0; nt < 2; nt++)
        acc[mt][nt] = __builtin_amdgcn_wmma_f32_16x16x32_bf16(
            false, A[mt].v, false, Bf[nt].v, (short)0, acc[mt][nt], false, false);
    cp16_wait();
    __syncthreads();
  }

  #pragma unroll
  for (int mt = 0; mt < 2; mt++) {
    #pragma unroll
    for (int nt = 0; nt < 2; nt++) {
      int n = bn * TILE_N + nw * 32 + nt * 16 + nl;
      float bias = b_experts[e * D_DIM + n];
      #pragma unroll
      for (int r = 0; r < 8; r++) {
        int m_local = mw * 32 + mt * 16 + r + 8 * g;
        if (bm * BM + m_local < cnt) {
          int entry = lds_list[m_local];
          int tok = entry >> 1;
          float v = gelu_tanh(acc[mt][nt][r] + bias) * gates[entry];
          atomicAdd(&combined[(size_t)tok * D_DIM + n], v);
        }
      }
    }
  }
}

// ---------------- Output projection: combined(f32) @ w_out + b_out -----------
__global__ __launch_bounds__(256) void out_gemm_kernel(
    const float* __restrict__ combined, const __bf16* __restrict__ woutT,
    const float* __restrict__ b_out, float* __restrict__ out) {
  __shared__ __bf16 lds_a[2][BM * LDA];
  __shared__ __bf16 lds_b[2][TILE_N * LDB];

  int bn = blockIdx.x, bm = blockIdx.y;
  int tid = threadIdx.x;
  const __bf16* Wn = woutT + (size_t)(bn * TILE_N) * D_DIM;

  // A staging: f32 source, cvt+pack (VGPR path); B staging: async DMA
  int t8r = tid >> 3;            // 0..31
  int t8c = (tid & 7) * 4;       // 0..28 (f32 elems)
  const float* cptr0 = combined + (size_t)(bm * BM + t8r) * D_DIM + t8c;
  const float* cptr1 = cptr0 + (size_t)32 * D_DIM;
  unsigned aoff0 = (unsigned)(t8r * LDA + t8c);
  unsigned aoff1 = aoff0 + 32u * LDA;

  int t4r = tid >> 2;
  int t4c = (tid & 3) * 8;
  const __bf16* bptr0 = Wn + (size_t)t4r * D_DIM + t4c;
  const __bf16* bptr1 = Wn + (size_t)(t4r + 64) * D_DIM + t4c;
  unsigned boff0 = (unsigned)(t4r * LDB + t4c);
  unsigned boff1 = (unsigned)((t4r + 64) * LDB + t4c);

  auto stage = [&](int buf, int k0) {
    cp16_g2l(bptr0 + k0, &lds_b[buf][boff0]);
    cp16_g2l(bptr1 + k0, &lds_b[buf][boff1]);
    float4 v0 = *(const float4*)(cptr0 + k0);
    float4 v1 = *(const float4*)(cptr1 + k0);
    uint2 p0 = { pack_bf16(v0.x, v0.y), pack_bf16(v0.z, v0.w) };
    uint2 p1 = { pack_bf16(v1.x, v1.y), pack_bf16(v1.z, v1.w) };
    *(uint2*)&lds_a[buf][aoff0] = p0;
    *(uint2*)&lds_a[buf][aoff1] = p1;
  };

  int wave = tid >> 5, lane = tid & 31;
  int mw = wave >> 2, nw = wave & 3;
  int g = lane >> 4, nl = lane & 15;

  v8f acc[2][2] = {};
  stage(0, 0);
  cp16_wait();
  __syncthreads();

  const int NSTEPS = D_DIM / KSTEP;
  for (int s = 0; s < NSTEPS; s++) {
    int cur = s & 1;
    if (s + 1 < NSTEPS) stage(cur ^ 1, (s + 1) * KSTEP);

    const __bf16* ab = lds_a[cur];
    const __bf16* bb = lds_b[cur];
    union { v16bf v; unsigned u[8]; } A[2], Bf[2];
    #pragma unroll
    for (int mt = 0; mt < 2; mt++) {
      int m = mw * 32 + mt * 16 + nl;
      const unsigned* alo = (const unsigned*)&ab[m * LDA + g * 8];
      const unsigned* ahi = (const unsigned*)&ab[m * LDA + 16 + g * 8];
      #pragma unroll
      for (int j = 0; j < 4; j++) { A[mt].u[j] = alo[j]; A[mt].u[4 + j] = ahi[j]; }
    }
    #pragma unroll
    for (int nt = 0; nt < 2; nt++) {
      int n = nw * 32 + nt * 16 + nl;
      const unsigned* blo = (const unsigned*)&bb[n * LDB + g * 8];
      const unsigned* bhi = (const unsigned*)&bb[n * LDB + 16 + g * 8];
      #pragma unroll
      for (int j = 0; j < 4; j++) { Bf[nt].u[j] = blo[j]; Bf[nt].u[4 + j] = bhi[j]; }
    }
    #pragma unroll
    for (int mt = 0; mt < 2; mt++)
      #pragma unroll
      for (int nt = 0; nt < 2; nt++)
        acc[mt][nt] = __builtin_amdgcn_wmma_f32_16x16x32_bf16(
            false, A[mt].v, false, Bf[nt].v, (short)0, acc[mt][nt], false, false);
    cp16_wait();
    __syncthreads();
  }

  #pragma unroll
  for (int mt = 0; mt < 2; mt++) {
    #pragma unroll
    for (int nt = 0; nt < 2; nt++) {
      int n = bn * TILE_N + nw * 32 + nt * 16 + nl;
      float bias = b_out[n];
      #pragma unroll
      for (int r = 0; r < 8; r++) {
        int m_local = mw * 32 + mt * 16 + r + 8 * g;
        out[(size_t)(bm * BM + m_local) * D_DIM + n] = acc[mt][nt][r] + bias;
      }
    }
  }
}

extern "C" void kernel_launch(void* const* d_in, const int* in_sizes, int n_in,
                              void* d_out, int out_size, void* d_ws, size_t ws_size,
                              hipStream_t stream) {
  const float* x         = (const float*)d_in[0];
  const float* w_router  = (const float*)d_in[1];
  const float* b_router  = (const float*)d_in[2];
  const float* w_experts = (const float*)d_in[3];
  const float* b_experts = (const float*)d_in[4];
  const float* w_out     = (const float*)d_in[5];
  const float* b_out     = (const float*)d_in[6];
  float* out = (float*)d_out;

  char* ws = (char*)d_ws;
  size_t off = 0;
  float* combined = (float*)(ws + off); off += (size_t)NTOK * D_DIM * sizeof(float); // 32MB
  int*   counts   = (int*)(ws + off);   off += 256;
  float* gates    = (float*)(ws + off); off += (size_t)NTOK * 2 * sizeof(float);
  int*   lists    = (int*)(ws + off);   off += (size_t)NEXP * NTOK * sizeof(int);
  __bf16* xbf     = (__bf16*)(ws + off); off += (size_t)NTOK * H_DIM * 2;            // 16MB
  __bf16* wT      = (__bf16*)(ws + off); off += (size_t)NEXP * H_DIM * D_DIM * 2;    // 16MB
  __bf16* woutT   = (__bf16*)(ws + off); off += (size_t)D_DIM * D_DIM * 2;           // 2MB

  (void)hipMemsetAsync(d_ws, 0, (size_t)NTOK * D_DIM * sizeof(float) + 256, stream);

  cvt_x_kernel<<<(NTOK * H_DIM / 2) / 256, 256, 0, stream>>>(x, (unsigned*)xbf,
                                                             NTOK * H_DIM / 2);
  dim3 tg(D_DIM / 32, H_DIM / 32, NEXP);
  transpose_cvt_kernel<<<tg, 256, 0, stream>>>(w_experts, wT, H_DIM, D_DIM);
  dim3 tw(D_DIM / 32, D_DIM / 32, 1);
  transpose_cvt_kernel<<<tw, 256, 0, stream>>>(w_out, woutT, D_DIM, D_DIM);

  router_kernel<<<NTOK / 8, 256, 0, stream>>>(x, w_router, b_router,
                                              gates, counts, lists);

  dim3 eg(D_DIM / TILE_N, NTOK / BM, NEXP);
  expert_kernel<<<eg, 256, 0, stream>>>(xbf, wT, b_experts, gates, counts, lists,
                                        combined);

  dim3 og(D_DIM / TILE_N, NTOK / BM, 1);
  out_gemm_kernel<<<og, 256, 0, stream>>>(combined, woutT, b_out, out);
}
